// AttentionShareLocal_8461085573553
// MI455X (gfx1250) — matrix operand order repaired and checked by reading, not compiled, not run on hardware
//
#include <hip/hip_runtime.h>

typedef __attribute__((ext_vector_type(16))) _Float16 v16h;
typedef __attribute__((ext_vector_type(8)))  float    v8f;

#define GS    7
#define NTOK  49
#define NH    8
#define CDIM  256
#define STR   72    // f16 LDS row stride (144B: 16B aligned, bank-friendly)
#define SROW  65    // f32 LDS row stride for scores (conflict-free column walks)

union HVec { v16h v; uint4 u[2]; };

// ---------------------------------------------------------------------------
// Kernel 1: relative-position-bias MLP  (169,2) -> relu(64) -> (169,8)
// ---------------------------------------------------------------------------
__global__ __launch_bounds__(256) void bias_mlp_kernel(
    const float* __restrict__ W1, const float* __restrict__ b1,
    const float* __restrict__ W2, const float* __restrict__ b2,
    float* __restrict__ pos_g) {
  int t = threadIdx.x;
  if (t < (2 * GS - 1) * (2 * GS - 1)) {   // 169 offsets
    float bh = (float)(t / (2 * GS - 1) - (GS - 1));
    float bw = (float)(t % (2 * GS - 1) - (GS - 1));
    float acc[NH];
#pragma unroll
    for (int hd = 0; hd < NH; ++hd) acc[hd] = b2[hd];
    for (int kk = 0; kk < 64; ++kk) {
      float hv = bh * W1[kk] + bw * W1[64 + kk] + b1[kk];
      hv = hv > 0.f ? hv : 0.f;                       // relu
#pragma unroll
      for (int hd = 0; hd < NH; ++hd) acc[hd] += hv * W2[kk * NH + hd];
    }
#pragma unroll
    for (int hd = 0; hd < NH; ++hd) pos_g[t * NH + hd] = acc[hd];
  }
}

// ---------------------------------------------------------------------------
// Kernel 2: fused windowed attention, one workgroup per (batch, head).
// 128 threads = 4 wave32s; each wave owns a 16-row strip of the 49(->64)
// token window. Both GEMMs run on v_wmma_f32_16x16x32_f16.
// ---------------------------------------------------------------------------
__global__ __launch_bounds__(128) void attn_kernel(
    const float* __restrict__ q, const float* __restrict__ k,
    const float* __restrict__ v, const float* __restrict__ pos_g,
    float* __restrict__ out) {
  __shared__ __align__(16) _Float16 Qh[64 * STR];   // Q * scale (also reused as P)
  __shared__ __align__(16) _Float16 Kh[64 * STR];   // K row-major [token][dim]
  __shared__ __align__(16) _Float16 VT[64 * STR];   // V transposed [dim][token]
  __shared__ float S[64 * SROW];                    // f32 scores / probs
  __shared__ float posl[(2 * GS - 1) * (2 * GS - 1) * NH];

  const int tid  = threadIdx.x;
  const int lane = tid & 31;
  const int wv   = tid >> 5;                 // wave id 0..3
  const int head = blockIdx.x & (NH - 1);
  const int bat  = blockIdx.x >> 3;

  const size_t base  = (size_t)bat * NTOK * CDIM + (size_t)head * 64;
  const float  scale = 0.125f;               // 64^-0.5, folded into Q

  // ---- stage q/k/v head tiles into LDS as f16; pad rows/cols 49..63 ----
  for (int idx = tid; idx < NTOK * 64; idx += 128) {
    int r = idx >> 6, c = idx & 63;
    size_t g = base + (size_t)r * CDIM + c;
    Qh[r * STR + c] = (_Float16)(q[g] * scale);
    Kh[r * STR + c] = (_Float16)(k[g]);
    VT[c * STR + r] = (_Float16)(v[g]);      // transpose V on the way in
  }
  for (int idx = tid; idx < (64 - NTOK) * 64; idx += 128) {
    int r = NTOK + (idx >> 6), c = idx & 63;
    Qh[r * STR + c] = (_Float16)0.f;
    Kh[r * STR + c] = (_Float16)0.f;
    VT[c * STR + r] = (_Float16)0.f;
  }
  for (int idx = tid; idx < 169 * NH; idx += 128) posl[idx] = pos_g[idx];
  __syncthreads();

  const int m16  = lane & 15;   // A row / B column / D column within tile
  const int hgrp = lane >> 4;   // lane half-group (ISA operand striping)
  const int arow = wv * 16 + m16;

  // ---- Phase A: S = (Q*scale) @ K^T ----
#pragma unroll
  for (int nt = 0; nt < 4; ++nt) {
    v8f c = {};
#pragma unroll
    for (int ks = 0; ks < 2; ++ks) {
      HVec a, b;
      int ka = ks * 32 + hgrp * 8;            // A: chunks [ka,ka+8) and +16
      a.u[0] = *(const uint4*)&Qh[arow * STR + ka];
      a.u[1] = *(const uint4*)&Qh[arow * STR + ka + 16];
      int kb = ks * 32 + hgrp * 8;            // B: K[j] chunks [kb,kb+8) and +16
      const _Float16* bp = &Kh[(nt * 16 + m16) * STR + kb];
      b.u[0] = *(const uint4*)(bp);
      b.u[1] = *(const uint4*)(bp + 16);
      c = __builtin_amdgcn_wmma_f32_16x16x32_f16(false, a.v, false, b.v,
                                                 (short)0, c, false, false);
    }
#pragma unroll
    for (int r = 0; r < 8; ++r) {             // D: VGPR r -> row r + 8*hgrp
      int m = wv * 16 + r + 8 * hgrp;
      S[m * SROW + nt * 16 + m16] = c[r];
    }
  }
  __syncthreads();

  // ---- softmax + relative position bias (f32, one thread per row) ----
  _Float16* P = Qh;                           // Q strip is dead; reuse as P
  if (tid < NTOK) {
    const int i = tid, ih = i / GS, iw = i % GS;
    float mx = -3.0e38f;
    for (int j = 0; j < NTOK; ++j) {
      int jh = j / GS, jw = j % GS;
      int rel = (ih - jh + GS - 1) * (2 * GS - 1) + (iw - jw + GS - 1);
      float val = S[i * SROW + j] + posl[rel * NH + head];
      S[i * SROW + j] = val;
      mx = fmaxf(mx, val);
    }
    float sum = 0.f;
    for (int j = 0; j < NTOK; ++j) {
      float e = __expf(S[i * SROW + j] - mx);
      S[i * SROW + j] = e;
      sum += e;
    }
    float inv = 1.f / sum;
    for (int j = 0; j < NTOK; ++j) P[i * STR + j] = (_Float16)(S[i * SROW + j] * inv);
    for (int j = NTOK; j < 64; ++j) P[i * STR + j] = (_Float16)0.f;
  } else if (tid < 64) {                      // zero padded probability rows
    for (int j = 0; j < 64; ++j) P[tid * STR + j] = (_Float16)0.f;
  }
  __syncthreads();

  // ---- Phase B: out = P @ V ----
#pragma unroll
  for (int nt = 0; nt < 4; ++nt) {
    v8f c = {};
#pragma unroll
    for (int ks = 0; ks < 2; ++ks) {
      HVec a, b;
      int ka = ks * 32 + hgrp * 8;
      a.u[0] = *(const uint4*)&P[arow * STR + ka];
      a.u[1] = *(const uint4*)&P[arow * STR + ka + 16];
      int kb = ks * 32 + hgrp * 8;            // B: VT[dim] chunks [kb,kb+8) and +16
      const _Float16* bp = &VT[(nt * 16 + m16) * STR + kb];
      b.u[0] = *(const uint4*)(bp);
      b.u[1] = *(const uint4*)(bp + 16);
      c = __builtin_amdgcn_wmma_f32_16x16x32_f16(false, a.v, false, b.v,
                                                 (short)0, c, false, false);
    }
#pragma unroll
    for (int r = 0; r < 8; ++r) {
      int m = wv * 16 + r + 8 * hgrp;
      if (m < NTOK)
        out[base + (size_t)m * CDIM + nt * 16 + m16] = c[r];
    }
  }
}

// ---------------------------------------------------------------------------
extern "C" void kernel_launch(void* const* d_in, const int* in_sizes, int n_in,
                              void* d_out, int out_size, void* d_ws, size_t ws_size,
                              hipStream_t stream) {
  (void)n_in; (void)out_size; (void)ws_size;
  const float* q  = (const float*)d_in[0];
  const float* k  = (const float*)d_in[1];
  const float* v  = (const float*)d_in[2];
  const float* W1 = (const float*)d_in[3];
  const float* b1 = (const float*)d_in[4];
  const float* W2 = (const float*)d_in[5];
  const float* b2 = (const float*)d_in[6];
  // d_in[7]=H, d_in[8]=W: with N=49 both reference branches add bias to the
  // full 49x49 window, so they are not needed.
  float* out = (float*)d_out;
  float* pos = (float*)d_ws;                  // 169*8 floats of scratch

  bias_mlp_kernel<<<1, 256, 0, stream>>>(W1, b1, W2, b2, pos);

  const int B = in_sizes[0] / (NTOK * CDIM);  // 2048 windows
  attn_kernel<<<B * NH, 128, 0, stream>>>(q, k, v, pos, out);
}
